// EBFlowHead_91199335563837
// MI455X (gfx1250) — matrix operand order repaired and checked by reading, not compiled, not run on hardware
//
#include <hip/hip_runtime.h>
#include <math.h>

// ---------------------------------------------------------------------------
// EBFlowHead for MI455X (gfx1250, wave32, WMMA).
// Dominant cost: per-pixel coupling MLPs (~94 GFLOP) -> fused LDS-resident
// f16 WMMA GEMM kernel (v_wmma_f32_16x16x32_f16), weights DMA'd into LDS via
// global_load_async_to_lds_b128 (ASYNCcnt). Everything else is
// bandwidth-trivial VALU work.
// ---------------------------------------------------------------------------

typedef _Float16 v16h __attribute__((ext_vector_type(16)));
typedef float    v8f  __attribute__((ext_vector_type(8)));
typedef _Float16 h8   __attribute__((ext_vector_type(8)));

#define BATCH 4
#define IMH 192
#define IMW 192
#define IMHW (IMH * IMW)
#define NPIX (BATCH * IMHW)      // 147456
#define PI_F 3.14159265358979f

__constant__ int c_PERM[6][3]  = {{1,2,0},{2,0,1},{0,2,1},{1,0,2},{2,1,0},{1,2,0}};
__constant__ int c_IPERM[6][3] = {{2,0,1},{1,2,0},{0,2,1},{1,0,2},{2,1,0},{2,0,1}};

// acc slots
#define ACC_NLL   0
#define ACC_MEANW 1
#define ACC_SSIM  2
#define ACC_TVX   3
#define ACC_TVY   4
#define ACC_PWR   5
#define ACC_ANLS  6

// ------------------------------ helpers ------------------------------------

__device__ __forceinline__ float geluf(float x) {
  return 0.5f * x * (1.f + erff(x * 0.70710678118654752f));
}
__device__ __forceinline__ float softplusf(float x) {
  return x > 20.f ? x : log1pf(expf(x));
}
__device__ __forceinline__ float block_reduce_add(float v, float* red) {
  int tid = threadIdx.x;
  red[tid] = v;
  __syncthreads();
  for (int s = blockDim.x >> 1; s > 0; s >>= 1) {
    if (tid < s) red[tid] += red[tid + s];
    __syncthreads();
  }
  float r = red[0];
  __syncthreads();
  return r;
}

// ---------------- async global->LDS weight staging (CDNA5) ------------------
// Per ISA 15.18.3: each lane copies memory at SADDR+VADDR into LDS at its VDST address,
// 16 bytes per lane per issue, tracked with ASYNCcnt.
__device__ __forceinline__ void async_copy_to_lds(const void* g, void* l,
                                                  int bytes, int tid) {
  unsigned long long gaddr = (unsigned long long)(uintptr_t)g;
  unsigned lbase = (unsigned)(uintptr_t)l;   // low 32 bits = LDS offset
  for (int o = tid * 16; o < bytes; o += 256 * 16) {
    unsigned laddr = lbase + (unsigned)o;
    unsigned voff = (unsigned)o;
    asm volatile("global_load_async_to_lds_b128 %0, %1, %2"
                 :: "v"(laddr), "v"(voff), "s"(gaddr) : "memory");
  }
}
__device__ __forceinline__ void wait_async_lds() {
  asm volatile("s_wait_asynccnt 0x0" ::: "memory");
}

// ------------------------- WMMA block GEMM ---------------------------------
// A: LDS [128][128] f16 (lda=128).  Wl: LDS [Npad][ldw] f16 row-major
// ([out][in], so B-fragment columns are contiguous K runs).
// out = act(A @ Wl^T + biasLds); 8 waves, wave w owns rows [16w,16w+16).
// biasLds is padded in LDS -> unconditional read, straight-line writeback.
template <bool F16OUT, bool RELU>
__device__ __forceinline__ void gemm_block(
    const _Float16* __restrict__ A, const _Float16* __restrict__ Wl, int ldw,
    const float* __restrict__ biasLds, int ksteps, int ntiles,
    void* __restrict__ outp, int ldo, int tid) {
  const int lane  = tid & 31;
  const int wave  = tid >> 5;
  const int half  = lane >> 4;
  const int l16   = lane & 15;
  const int mbase = wave * 16;
  for (int nt = 0; nt < ntiles; ++nt) {
    const int ncol = nt * 16 + l16;
    const float bv = biasLds[ncol];
    v8f acc = {};
    for (int ks = 0; ks < ksteps; ++ks) {
      // A fragment: M = lane%16; element e -> K = (e>>3)*16 + half*8 + (e&7)
      const _Float16* ar = A + (mbase + l16) * 128 + ks * 32 + half * 8;
      h8 a0 = *(const h8*)(ar);
      h8 a1 = *(const h8*)(ar + 16);
      // B fragment: N = lane%16; element e -> K = (e>>3)*16 + half*8 + (e&7)
      const _Float16* br = Wl + ncol * ldw + ks * 32 + half * 8;
      h8 b0v = *(const h8*)(br);
      h8 b1v = *(const h8*)(br + 16);
      v16h va, vb;
#pragma unroll
      for (int e = 0; e < 8; ++e) {
        va[e] = a0[e]; va[e + 8] = a1[e];
        vb[e] = b0v[e]; vb[e + 8] = b1v[e];
      }
      acc = __builtin_amdgcn_wmma_f32_16x16x32_f16(
          false, va, false, vb, (short)0, acc, false, false);
    }
#pragma unroll
    for (int r = 0; r < 8; ++r) {
      int mm = mbase + r + 8 * half;   // C layout: M = r + 8*(lane>>4)
      float v = acc[r] + bv;
      if (RELU) v = fmaxf(v, 0.f);
      if (F16OUT) ((_Float16*)outp)[mm * ldo + ncol] = (_Float16)v;
      else        ((float*)outp)[mm * ldo + ncol] = v;
    }
  }
}

// -------------------------- RQ spline --------------------------------------

__device__ __forceinline__ void knots8(const float* __restrict__ u,
                                       float* c, float* w) {
  float mx = u[0];
#pragma unroll
  for (int j = 1; j < 8; ++j) mx = fmaxf(mx, u[j]);
  float e[8]; float s = 0.f;
#pragma unroll
  for (int j = 0; j < 8; ++j) { e[j] = expf(u[j] - mx); s += e[j]; }
  float inv = 1.f / s, cum = 0.f;
  c[0] = -3.f;
#pragma unroll
  for (int j = 0; j < 8; ++j) {
    float wj = 0.001f + 0.992f * e[j] * inv;   // MIN_W + (1-K*MIN_W)*softmax
    cum += wj;
    c[j + 1] = 6.f * cum - 3.f;
  }
  c[8] = 3.f;
#pragma unroll
  for (int j = 0; j < 8; ++j) w[j] = c[j + 1] - c[j];
}

template <int INV>
__device__ void rq_spline(const float* __restrict__ u, float xval,
                          float& outv, float& ladv) {
  const bool inside = (xval >= -3.f) && (xval <= 3.f);
  const float xc = fminf(fmaxf(xval, -3.f), 3.f);
  float cw[9], wk[8], ch[9], hk[8], dk[9];
  knots8(u, cw, wk);
  knots8(u + 8, ch, hk);
  const float DC = logf(expf(0.999f) - 1.f);
#pragma unroll
  for (int j = 0; j < 9; ++j) {
    float pv = (j == 0 || j == 8) ? DC : u[15 + j];
    dk[j] = 0.001f + softplusf(pv);
  }
  const float* loc = INV ? ch : cw;
  int idx = -1;
#pragma unroll
  for (int j = 0; j < 9; ++j) idx += (xc >= loc[j]) ? 1 : 0;
  idx = max(0, min(idx, 7));
  // unrolled select -> cndmask chain, avoids scratch for dynamic indexing
  float CW = cw[0], Wd = wk[0], CH = ch[0], Hh = hk[0], D0 = dk[0], D1 = dk[1];
#pragma unroll
  for (int j = 1; j < 8; ++j) {
    bool sel = (idx == j);
    CW = sel ? cw[j] : CW; Wd = sel ? wk[j] : Wd;
    CH = sel ? ch[j] : CH; Hh = sel ? hk[j] : Hh;
    D0 = sel ? dk[j] : D0; D1 = sel ? dk[j + 1] : D1;
  }
  float sk = Hh / Wd;
  float theta;
  if (INV) {
    float t  = xc - CH;
    float m2 = D0 + D1 - 2.f * sk;
    float a  = t * m2 + Hh * (sk - D0);
    float b  = Hh * D0 - t * m2;
    float c  = -sk * t;
    float disc = fmaxf(b * b - 4.f * a * c, 0.f);
    theta = 2.f * c / (-b - sqrtf(disc));
    outv = theta * Wd + CW;
  } else {
    theta = (xc - CW) / Wd;
    float omt = 1.f - theta;
    float num = Hh * (sk * theta * theta + D0 * theta * omt);
    float den = sk + (D0 + D1 - 2.f * sk) * theta * omt;
    outv = CH + num / den;
  }
  float omt = 1.f - theta;
  float den = sk + (D0 + D1 - 2.f * sk) * theta * omt;
  float lad = logf(sk * sk * (D1 * theta * theta + 2.f * sk * theta * omt +
                              D0 * omt * omt)) - 2.f * logf(den);
  if (INV) lad = -lad;
  if (!inside) { outv = xval; lad = 0.f; }
  ladv = lad;
}

// ------------------------------ kernels ------------------------------------

__global__ void zero_kernel(float* p, int n) {
  int i = blockIdx.x * blockDim.x + threadIdx.x;
  if (i < n) p[i] = 0.f;
}

// pad/convert weights to f16 [Npad][Kpad] (src is [N][K] row-major)
__global__ void prep_w_kernel(const float* __restrict__ src,
                              _Float16* __restrict__ dst,
                              int N, int K, int Npad, int Kpad) {
  int i = blockIdx.x * blockDim.x + threadIdx.x;
  if (i >= Npad * Kpad) return;
  int n = i / Kpad, k = i % Kpad;
  dst[i] = (n < N && k < K) ? (_Float16)src[n * K + k] : (_Float16)0.f;
}

__global__ void conv1_kernel(const float* __restrict__ feat,
                             const float* __restrict__ wt,
                             const float* __restrict__ bias,
                             float* __restrict__ out) {
  int idx = blockIdx.x * blockDim.x + threadIdx.x;
  const int CO = 64, CI = 10;
  if (idx >= BATCH * CO * IMHW) return;
  int x = idx % IMW, y = (idx / IMW) % IMH;
  int co = (idx / IMHW) % CO, b = idx / (IMHW * CO);
  float s = bias[co];
  for (int ci = 0; ci < CI; ++ci)
    for (int ky = 0; ky < 3; ++ky) {
      int yy = y + ky - 1;
      if (yy < 0 || yy >= IMH) continue;
      for (int kx = 0; kx < 3; ++kx) {
        int xx = x + kx - 1;
        if (xx < 0 || xx >= IMW) continue;
        float v;
        if (ci < 8)       v = feat[((b * 8 + ci) * IMH + yy) * IMW + xx];
        else if (ci == 8) v = -1.f + 2.f * xx / (IMW - 1);   // xx coord
        else              v = -1.f + 2.f * yy / (IMH - 1);   // yy coord
        s += v * wt[((co * CI + ci) * 3 + ky) * 3 + kx];
      }
    }
  out[idx] = geluf(s);
}

__global__ void conv2_kernel(const float* __restrict__ h1,
                             const float* __restrict__ wt,
                             const float* __restrict__ bias,
                             float* __restrict__ out) {
  int idx = blockIdx.x * blockDim.x + threadIdx.x;
  const int CO = 32, CI = 64;
  if (idx >= BATCH * CO * IMHW) return;
  int x = idx % IMW, y = (idx / IMW) % IMH;
  int co = (idx / IMHW) % CO, b = idx / (IMHW * CO);
  float s = bias[co];
  for (int ci = 0; ci < CI; ++ci) {
    const float* src = h1 + ((size_t)(b * CI + ci)) * IMHW;
    const float* wr  = wt + ((co * CI + ci) * 9);
    for (int ky = 0; ky < 3; ++ky) {
      int yy = y + ky - 1;
      if (yy < 0 || yy >= IMH) continue;
      for (int kx = 0; kx < 3; ++kx) {
        int xx = x + kx - 1;
        if (xx < 0 || xx >= IMW) continue;
        s += src[yy * IMW + xx] * wr[ky * 3 + kx];
      }
    }
  }
  out[idx] = geluf(s);
}

__global__ void stats_kernel(const float* __restrict__ h2,
                             float* __restrict__ stats) {
  __shared__ float red[256];
  int bc = blockIdx.x;                 // (b*32 + c)
  const float* src = h2 + (size_t)bc * IMHW;
  float s = 0.f, s2 = 0.f;
  for (int i = threadIdx.x; i < IMHW; i += 256) {
    float v = src[i]; s += v; s2 += v * v;
  }
  float S = block_reduce_add(s, red);
  float S2 = block_reduce_add(s2, red);
  if (threadIdx.x == 0) {
    float mean = S / IMHW;
    float var = S2 / IMHW - mean * mean;
    stats[bc * 2]     = mean;
    stats[bc * 2 + 1] = rsqrtf(var + 1e-5f);
  }
}

__global__ void ctx_kernel(const float* __restrict__ h2,
                           const float* __restrict__ stats,
                           _Float16* __restrict__ ctx16) {
  int m = blockIdx.x * blockDim.x + threadIdx.x;
  if (m >= NPIX) return;
  int b = m / IMHW, hw = m % IMHW;
  for (int c = 0; c < 32; ++c) {
    float v = h2[((size_t)(b * 32 + c)) * IMHW + hw];
    float mean = stats[(b * 32 + c) * 2], istd = stats[(b * 32 + c) * 2 + 1];
    ctx16[(size_t)m * 32 + c] = (_Float16)((v - mean) * istd);
  }
}

__global__ void initfwd_kernel(const float* __restrict__ target,
                               float* __restrict__ xA,
                               float* __restrict__ ladg) {
  int i = blockIdx.x * blockDim.x + threadIdx.x;
  if (i >= NPIX * 3) return;
  int m = i / 3, c = i % 3;
  int b = m / IMHW, hw = m % IMHW;
  xA[i] = target[((size_t)(b * 3 + c)) * IMHW + hw];  // NCHW -> [M][3]
  if (i < NPIX) ladg[i] = 0.f;
}

__global__ void initinv_kernel(float* __restrict__ xA) {
  int i = blockIdx.x * blockDim.x + threadIdx.x;
  if (i < NPIX * 3) xA[i] = 0.f;
}

// ----------- fused coupling layer: affine + MLP (WMMA) + spline ------------
template <int INV>
__global__ void __launch_bounds__(256)
coupling_kernel(const float* __restrict__ xin, float* __restrict__ xout,
                float* __restrict__ ladg, const _Float16* __restrict__ ctx16,
                const _Float16* __restrict__ w0g,
                const _Float16* __restrict__ w1g,
                const _Float16* __restrict__ w2g,
                const float* __restrict__ b0, const float* __restrict__ b1,
                const float* __restrict__ b2, const float* __restrict__ an_ls,
                const float* __restrict__ an_sh, int li) {
  extern __shared__ char smem[];
  _Float16* Sa  = (_Float16*)smem;         // [128][128]
  _Float16* Sb  = Sa + 128 * 128;          // [128][128]
  _Float16* W0l = Sb + 128 * 128;          // [128][64]
  _Float16* W1l = W0l + 128 * 64;          // [128][128]
  _Float16* W2l = W1l + 128 * 128;         // up to [48][128]
  float* Pb  = (float*)(W2l + 48 * 128);   // [128][48] params
  float* Yb  = Pb + 128 * 48;              // [128][3]
  float* Lb  = Yb + 128 * 3;               // [128][2]
  float* Bb0 = Lb + 128 * 2;               // [128]
  float* Bb1 = Bb0 + 128;                  // [128]
  float* Bb2 = Bb1 + 128;                  // [48] zero padded

  const int tid = threadIdx.x;
  const int m0 = blockIdx.x * 128;
  const bool even = (li & 1) == 0;
  const int d_id = even ? 1 : 2;
  const int n_tr = even ? 2 : 1;
  const int NOUT2 = n_tr * 23;
  const int NT2 = even ? 3 : 2;
  const float* ls = an_ls + li * 3;
  const float* sh = an_sh + li * 3;

  // ---- async DMA all three weight matrices into LDS (overlaps A build) ----
  async_copy_to_lds(w0g, W0l, 128 * 64 * 2, tid);
  async_copy_to_lds(w1g, W1l, 128 * 128 * 2, tid);
  async_copy_to_lds(w2g, W2l, (even ? 48 : 32) * 128 * 2, tid);

  // ---- biases into LDS (padded -> unconditional reads in GEMM) ----
  if (tid < 128) { Bb0[tid] = b0[tid]; Bb1[tid] = b1[tid]; }
  if (tid < 48)  Bb2[tid] = (tid < NOUT2) ? b2[tid] : 0.f;

  // ---- build layer-0 input A in Sa (ident channels + f16 ctx, zero pad) ---
  {
    int row = tid >> 1;
    int kh = tid & 1;
    int m = m0 + row;
    float id0v = 0.f, id1v = 0.f;
    if (kh == 0) {
      if (!INV) {
        if (even) { id0v = expf(ls[1]) * xin[m * 3 + 1] + sh[1]; }
        else {
          id0v = expf(ls[0]) * xin[m * 3 + 0] + sh[0];
          id1v = expf(ls[2]) * xin[m * 3 + 2] + sh[2];
        }
      } else {
        const int* ip = c_IPERM[li];
        if (even) { id0v = xin[m * 3 + ip[1]]; }
        else { id0v = xin[m * 3 + ip[0]]; id1v = xin[m * 3 + ip[2]]; }
      }
    }
    int kbase = kh * 32;
    for (int kk = 0; kk < 32; ++kk) {
      int k = kbase + kk;
      _Float16 v;
      if (k < d_id)           v = (_Float16)(k == 0 ? id0v : id1v);
      else if (k < d_id + 32) v = ctx16[(size_t)m * 32 + (k - d_id)];
      else                    v = (_Float16)0.f;
      Sa[row * 128 + k] = v;
    }
  }
  wait_async_lds();
  __syncthreads();

  gemm_block<true, true>(Sa, W0l, 64, Bb0, 2, 8, Sb, 128, tid);
  __syncthreads();
  gemm_block<true, true>(Sb, W1l, 128, Bb1, 4, 8, Sa, 128, tid);
  __syncthreads();
  gemm_block<false, false>(Sa, W2l, 128, Bb2, 4, NT2, Pb, 48, tid);
  __syncthreads();

  // ---- spline on transformed channels ----
  if (tid < 128 * n_tr) {
    int row = tid & 127;
    int t = tid >> 7;
    int m = m0 + row;
    int ch = even ? (t == 0 ? 0 : 2) : 1;
    float xv;
    if (!INV) {
      xv = expf(ls[ch]) * xin[m * 3 + ch] + sh[ch];
      if (t == 0) {
        if (even) Yb[row * 3 + 1] = expf(ls[1]) * xin[m * 3 + 1] + sh[1];
        else {
          Yb[row * 3 + 0] = expf(ls[0]) * xin[m * 3 + 0] + sh[0];
          Yb[row * 3 + 2] = expf(ls[2]) * xin[m * 3 + 2] + sh[2];
        }
      }
    } else {
      const int* ip = c_IPERM[li];
      xv = xin[m * 3 + ip[ch]];
      if (t == 0) {
        if (even) Yb[row * 3 + 1] = xin[m * 3 + ip[1]];
        else {
          Yb[row * 3 + 0] = xin[m * 3 + ip[0]];
          Yb[row * 3 + 2] = xin[m * 3 + ip[2]];
        }
      }
    }
    float outv, ladv;
    rq_spline<INV>(&Pb[row * 48 + t * 23], xv, outv, ladv);
    Yb[row * 3 + ch] = outv;
    Lb[row * 2 + t] = ladv;
  }
  __syncthreads();

  // ---- permute + write (fwd) / inverse affine + write (inv) ----
  for (int t = tid; t < 128 * 3; t += 256) {
    int row = t / 3, j = t % 3;
    int m = m0 + row;
    if (!INV) {
      xout[m * 3 + j] = Yb[row * 3 + c_PERM[li][j]];
      if (j == 0)
        ladg[m] += Lb[row * 2] + (n_tr > 1 ? Lb[row * 2 + 1] : 0.f);
    } else {
      xout[m * 3 + j] = (Yb[row * 3 + j] - sh[j]) * expf(-ls[j]);
    }
  }
}

// scalar precompute: meanW of cosine window, sum(an_ls)
__global__ void scalars_kernel(float* __restrict__ acc,
                               const float* __restrict__ an_ls) {
  if (threadIdx.x != 0 || blockIdx.x != 0) return;
  float sy = 0.f, sx = 0.f;
  for (int h = 0; h < IMH; ++h) sy += 0.5f * (1.f - cosf(PI_F * h / (IMH - 1)));
  for (int w = 0; w < IMW; ++w) sx += 0.5f * (1.f - cosf(PI_F * w / (IMW - 1)));
  acc[ACC_MEANW] = 1.f - (sy / IMH) * (sx / IMW);
  float s = 0.f;
  for (int i = 0; i < 18; ++i) s += an_ls[i];
  acc[ACC_ANLS] = s;
}

__global__ void logprob_kernel(const float* __restrict__ x,
                               const float* __restrict__ ladg,
                               float* __restrict__ acc) {
  __shared__ float red[256];
  int m = blockIdx.x * blockDim.x + threadIdx.x;
  float partial = 0.f;
  if (m < NPIX) {
    int hw = m % IMHW, h = hw / IMW, w = hw % IMW;
    float wy = 0.5f * (1.f - cosf(PI_F * h / (IMH - 1)));
    float wx = 0.5f * (1.f - cosf(PI_F * w / (IMW - 1)));
    float raw = 1.f - wy * wx;
    float x0 = x[m * 3], x1 = x[m * 3 + 1], x2 = x[m * 3 + 2];
    float lp = -0.5f * (x0 * x0 + x1 * x1 + x2 * x2) - 2.7568155996140186f +
               ladg[m] + acc[ACC_ANLS];
    partial = lp * raw;
  }
  float s = block_reduce_add(partial, red);
  if (threadIdx.x == 0) atomicAdd(&acc[ACC_NLL], s);
}

__global__ void sample_kernel(const float* __restrict__ xinv,
                              const float* __restrict__ target,
                              const float* __restrict__ acc,
                              float* __restrict__ sample,
                              float* __restrict__ sw,
                              float* __restrict__ tw) {
  int i = blockIdx.x * blockDim.x + threadIdx.x;   // [B][3][HW]
  if (i >= NPIX * 3) return;
  int hw = i % IMHW, c = (i / IMHW) % 3, b = i / (IMHW * 3);
  int h = hw / IMW, w = hw % IMW;
  int m = b * IMHW + hw;
  float wy = 0.5f * (1.f - cosf(PI_F * h / (IMH - 1)));
  float wx = 0.5f * (1.f - cosf(PI_F * w / (IMW - 1)));
  float win = (1.f - wy * wx) / acc[ACC_MEANW];
  float s = xinv[m * 3 + c];
  sample[i] = s;
  sw[i] = s * win;
  tw[i] = target[i] * win;
}

__global__ void tv_kernel(const float* __restrict__ sample,
                          float* __restrict__ acc) {
  __shared__ float red[256];
  int i = blockIdx.x * blockDim.x + threadIdx.x;
  float dx = 0.f, dy = 0.f;
  if (i < NPIX * 3) {
    int w = i % IMW, h = (i / IMW) % IMH;
    float v = sample[i];
    if (w < IMW - 1) dx = fabsf(sample[i + 1] - v);
    if (h < IMH - 1) dy = fabsf(sample[i + IMW] - v);
  }
  float sx = block_reduce_add(dx, red);
  float sy = block_reduce_add(dy, red);
  if (threadIdx.x == 0) {
    atomicAdd(&acc[ACC_TVX], sx);
    atomicAdd(&acc[ACC_TVY], sy);
  }
}

__device__ __forceinline__ float blur_src(const float* sw, const float* tw,
                                          size_t i, int mode) {
  float a = sw[i], b = tw[i];
  switch (mode) {
    case 0: return a;
    case 1: return b;
    case 2: return a * a;
    case 3: return b * b;
    default: return a * b;
  }
}

__global__ void blur_h_kernel(const float* __restrict__ sw,
                              const float* __restrict__ tw,
                              float* __restrict__ out, int mode) {
  int i = blockIdx.x * blockDim.x + threadIdx.x;
  if (i >= NPIX * 3) return;
  int x = i % IMW;
  size_t base = (size_t)i - x;
  float g[11], gs = 0.f;
#pragma unroll
  for (int j = 0; j < 11; ++j) { g[j] = expf(-(j - 5) * (j - 5) / 4.5f); gs += g[j]; }
  float inv = 1.f / gs;
  float s = 0.f;
#pragma unroll
  for (int j = 0; j < 11; ++j) {
    int xx = x + j - 5;
    if (xx >= 0 && xx < IMW) s += g[j] * blur_src(sw, tw, base + xx, mode);
  }
  out[i] = s * inv;
}

__global__ void blur_v_kernel(const float* __restrict__ in,
                              float* __restrict__ out) {
  int i = blockIdx.x * blockDim.x + threadIdx.x;
  if (i >= NPIX * 3) return;
  int x = i % IMW, y = (i / IMW) % IMH;
  size_t plane = (size_t)(i / IMHW) * IMHW;
  float g[11], gs = 0.f;
#pragma unroll
  for (int j = 0; j < 11; ++j) { g[j] = expf(-(j - 5) * (j - 5) / 4.5f); gs += g[j]; }
  float inv = 1.f / gs;
  float s = 0.f;
#pragma unroll
  for (int j = 0; j < 11; ++j) {
    int yy = y + j - 5;
    if (yy >= 0 && yy < IMH) s += g[j] * in[plane + yy * IMW + x];
  }
  out[i] = s * inv;
}

__global__ void ssim_kernel(const float* __restrict__ mp,
                            const float* __restrict__ mt,
                            const float* __restrict__ cpp,
                            const float* __restrict__ ctt,
                            const float* __restrict__ cpt,
                            float* __restrict__ acc) {
  __shared__ float red[256];
  int i = blockIdx.x * blockDim.x + threadIdx.x;
  float v = 0.f;
  if (i < NPIX * 3) {
    const float C1 = 1e-4f, C2 = 9e-4f;
    float a = mp[i], b = mt[i];
    float mpp = a * a, mtt = b * b, mpt = a * b;
    float sp = cpp[i] - mpp, st = ctt[i] - mtt, spt = cpt[i] - mpt;
    v = (2.f * mpt + C1) * (2.f * spt + C2) /
        ((mpp + mtt + C1) * (sp + st + C2));
  }
  float s = block_reduce_add(v, red);
  if (threadIdx.x == 0) atomicAdd(&acc[ACC_SSIM], s);
}

__device__ __forceinline__ int radial_bin(int ky, int kx) {
  float kyf = (ky < IMH / 2) ? (float)ky : (float)(ky - IMH);
  float kxf = (kx < IMW / 2) ? (float)kx : (float)(kx - IMW);
  int b = (int)rintf(sqrtf(kyf * kyf + kxf * kxf));
  return b > 95 ? 95 : b;
}

__global__ void counts_kernel(float* __restrict__ counts) {
  int i = blockIdx.x * blockDim.x + threadIdx.x;
  if (i >= IMHW) return;
  atomicAdd(&counts[radial_bin(i / IMW, i % IMW)], 1.f);
}

__global__ void dft_row_kernel(const float* __restrict__ sw,
                               const float* __restrict__ tw,
                               float* __restrict__ rowdft) {
  int i = blockIdx.x * blockDim.x + threadIdx.x;   // p*HW + y*W + kx
  if (i >= 24 * IMHW) return;
  int kx = i % IMW, y = (i / IMW) % IMH, p = i / IMHW;
  const float* src = ((p < 12) ? sw : tw) + (size_t)(p % 12) * IMHW + y * IMW;
  float ang = -2.f * PI_F * kx / IMW;
  float cr, ci;
  sincosf(ang, &ci, &cr);   // ci = sin, cr = cos
  float zr = 1.f, zi = 0.f, ar = 0.f, ai = 0.f;
  for (int x = 0; x < IMW; ++x) {
    float v = src[x];
    ar += v * zr; ai += v * zi;
    float nr = zr * cr - zi * ci;
    zi = zr * ci + zi * cr;
    zr = nr;
  }
  rowdft[(size_t)i * 2] = ar;
  rowdft[(size_t)i * 2 + 1] = ai;
}

__global__ void dft_col_kernel(const float* __restrict__ rowdft,
                               float* __restrict__ binsum) {
  int i = blockIdx.x * blockDim.x + threadIdx.x;   // p*HW + ky*W + kx
  if (i >= 24 * IMHW) return;
  int kx = i % IMW, ky = (i / IMW) % IMH, p = i / IMHW;
  float ang = -2.f * PI_F * ky / IMH;
  float cr, ci;
  sincosf(ang, &ci, &cr);
  float zr = 1.f, zi = 0.f, Fr = 0.f, Fi = 0.f;
  const float* base = rowdft + ((size_t)p * IMHW + kx) * 2;
  for (int y = 0; y < IMH; ++y) {
    float rr = base[(size_t)y * IMW * 2];
    float ri = base[(size_t)y * IMW * 2 + 1];
    Fr += rr * zr - ri * zi;
    Fi += rr * zi + ri * zr;
    float nr = zr * cr - zi * ci;
    zi = zr * ci + zi * cr;
    zr = nr;
  }
  float P = Fr * Fr + Fi * Fi;
  atomicAdd(&binsum[p * 96 + radial_bin(ky, kx)], P);
}

__global__ void pwr_kernel(const float* __restrict__ binsum,
                           const float* __restrict__ counts,
                           float* __restrict__ acc) {
  if (threadIdx.x != 0 || blockIdx.x != 0) return;
  float tot = 0.f;
  for (int c = 0; c < 3; ++c) {
    float s = 0.f;
    for (int b = 0; b < 4; ++b)
      for (int bin = 0; bin < 96; ++bin) {
        float g = binsum[(b * 3 + c) * 96 + bin] / counts[bin];
        float t = binsum[(12 + b * 3 + c) * 96 + bin] / counts[bin];
        s += fabsf(logf(g / (t + 1e-8f)));
      }
    tot += s / (4.f * 96.f);
  }
  acc[ACC_PWR] = tot;
}

__global__ void final_kernel(const float* __restrict__ acc,
                             float* __restrict__ out) {
  if (threadIdx.x != 0 || blockIdx.x != 0) return;
  float nll = -acc[ACC_NLL] / (acc[ACC_MEANW] * (float)NPIX);
  float ssim_loss = 1.f - acc[ACC_SSIM] / (float)(NPIX * 3);
  float tv = (acc[ACC_TVX] / (float)(BATCH * 3 * IMH * (IMW - 1)) +
              acc[ACC_TVY] / (float)(BATCH * 3 * (IMH - 1) * IMW)) * 1e-5f;
  out[0] = nll + 0.1f * ssim_loss + 0.05f * acc[ACC_PWR] + tv;
}

// ------------------------------ host ---------------------------------------

extern "C" void kernel_launch(void* const* d_in, const int* in_sizes, int n_in,
                              void* d_out, int out_size, void* d_ws,
                              size_t ws_size, hipStream_t stream) {
  const float* features = (const float*)d_in[0];
  const float* target   = (const float*)d_in[1];
  const float* cw1 = (const float*)d_in[2];
  const float* cb1 = (const float*)d_in[3];
  const float* cw2 = (const float*)d_in[4];
  const float* cb2 = (const float*)d_in[5];
  const float* an_ls = (const float*)d_in[6];
  const float* an_sh = (const float*)d_in[7];
  const float* ew0 = (const float*)d_in[8];
  const float* eb0 = (const float*)d_in[9];
  const float* ew1 = (const float*)d_in[10];
  const float* eb1 = (const float*)d_in[11];
  const float* ew2 = (const float*)d_in[12];
  const float* eb2 = (const float*)d_in[13];
  const float* ow0 = (const float*)d_in[14];
  const float* ob0 = (const float*)d_in[15];
  const float* ow1 = (const float*)d_in[16];
  const float* ob1 = (const float*)d_in[17];
  const float* ow2 = (const float*)d_in[18];
  const float* ob2 = (const float*)d_in[19];

  char* ws = (char*)d_ws;
  size_t off = 0;
  auto alloc = [&](size_t bytes) {
    size_t o = off;
    off += (bytes + 255) & ~(size_t)255;
    return o;
  };
  auto cdiv = [](int a, int b) { return (a + b - 1) / b; };

  const int M = NPIX;
  float* acc = (float*)(ws + alloc(2432 * sizeof(float)));
  float* counts = acc + 32;
  float* binsum = acc + 128;
  float* stats = (float*)(ws + alloc(BATCH * 32 * 2 * sizeof(float)));

  const int szE[3] = {128 * 64, 128 * 128, 48 * 128};
  const int szO[3] = {128 * 64, 128 * 128, 32 * 128};
  _Float16* wE[3][3];
  _Float16* wO[3][3];
  for (int s = 0; s < 3; ++s)
    for (int l = 0; l < 3; ++l) {
      wE[s][l] = (_Float16*)(ws + alloc((size_t)szE[l] * 2));
      wO[s][l] = (_Float16*)(ws + alloc((size_t)szO[l] * 2));
    }
  _Float16* ctx16 = (_Float16*)(ws + alloc((size_t)M * 32 * 2));
  float* xA   = (float*)(ws + alloc((size_t)M * 3 * 4));
  float* xB   = (float*)(ws + alloc((size_t)M * 3 * 4));
  float* ladg = (float*)(ws + alloc((size_t)M * 4));
  float* sample = (float*)(ws + alloc((size_t)M * 3 * 4));
  float* swb = (float*)(ws + alloc((size_t)M * 3 * 4));
  float* twb = (float*)(ws + alloc((size_t)M * 3 * 4));
  // big region: conv activations, later reused for DFT rows + blur buffers
  char* big = ws + alloc((size_t)BATCH * 96 * IMHW * 4);
  float* h1 = (float*)big;                       // [B][64][HW]
  float* h2 = h1 + (size_t)BATCH * 64 * IMHW;    // [B][32][HW]
  float* rowdft = (float*)big;                   // [24][HW][2]
  float* btmp = rowdft + (size_t)24 * IMHW * 2;  // [B][3][HW]
  float* mpA = btmp + (size_t)M * 3;
  float* mtA = mpA + (size_t)M * 3;
  float* cpp = mtA + (size_t)M * 3;
  float* ctt = cpp + (size_t)M * 3;
  float* cpt = ctt + (size_t)M * 3;
  float* blurOut[5] = {mpA, mtA, cpp, ctt, cpt};

  // LDS: Sa+Sb+W0+W1+W2 (f16) + params/Y/L/bias (f32)
  const int SMEM = (2 * 128 * 128 + 128 * 64 + 128 * 128 + 48 * 128) * 2 +
                   (128 * 48 + 128 * 3 + 128 * 2 + 128 + 128 + 48) * 4;

  zero_kernel<<<cdiv(2432, 256), 256, 0, stream>>>(acc, 2432);

  for (int s = 0; s < 3; ++s) {
    prep_w_kernel<<<cdiv(128 * 64, 256), 256, 0, stream>>>(
        ew0 + (size_t)s * 128 * 33, wE[s][0], 128, 33, 128, 64);
    prep_w_kernel<<<cdiv(128 * 128, 256), 256, 0, stream>>>(
        ew1 + (size_t)s * 128 * 128, wE[s][1], 128, 128, 128, 128);
    prep_w_kernel<<<cdiv(48 * 128, 256), 256, 0, stream>>>(
        ew2 + (size_t)s * 46 * 128, wE[s][2], 46, 128, 48, 128);
    prep_w_kernel<<<cdiv(128 * 64, 256), 256, 0, stream>>>(
        ow0 + (size_t)s * 128 * 34, wO[s][0], 128, 34, 128, 64);
    prep_w_kernel<<<cdiv(128 * 128, 256), 256, 0, stream>>>(
        ow1 + (size_t)s * 128 * 128, wO[s][1], 128, 128, 128, 128);
    prep_w_kernel<<<cdiv(32 * 128, 256), 256, 0, stream>>>(
        ow2 + (size_t)s * 23 * 128, wO[s][2], 23, 128, 32, 128);
  }

  conv1_kernel<<<cdiv(BATCH * 64 * IMHW, 256), 256, 0, stream>>>(
      features, cw1, cb1, h1);
  conv2_kernel<<<cdiv(BATCH * 32 * IMHW, 256), 256, 0, stream>>>(
      h1, cw2, cb2, h2);
  stats_kernel<<<BATCH * 32, 256, 0, stream>>>(h2, stats);
  ctx_kernel<<<cdiv(M, 256), 256, 0, stream>>>(h2, stats, ctx16);

  // ----- forward flow (logprob path) -----
  initfwd_kernel<<<cdiv(M * 3, 256), 256, 0, stream>>>(target, xA, ladg);
  {
    float* xi = xA; float* xo = xB;
    for (int li = 0; li < 6; ++li) {
      int s = li / 2;
      bool even = (li % 2 == 0);
      coupling_kernel<0><<<M / 128, 256, SMEM, stream>>>(
          xi, xo, ladg, ctx16,
          even ? wE[s][0] : wO[s][0], even ? wE[s][1] : wO[s][1],
          even ? wE[s][2] : wO[s][2],
          even ? (eb0 + s * 128) : (ob0 + s * 128),
          even ? (eb1 + s * 128) : (ob1 + s * 128),
          even ? (eb2 + s * 46) : (ob2 + s * 23),
          an_ls, an_sh, li);
      float* t = xi; xi = xo; xo = t;
    }
  }
  scalars_kernel<<<1, 1, 0, stream>>>(acc, an_ls);
  logprob_kernel<<<cdiv(M, 256), 256, 0, stream>>>(xA, ladg, acc);

  // ----- inverse flow (sample path) -----
  initinv_kernel<<<cdiv(M * 3, 256), 256, 0, stream>>>(xA);
  {
    float* xi = xA; float* xo = xB;
    for (int li = 5; li >= 0; --li) {
      int s = li / 2;
      bool even = (li % 2 == 0);
      coupling_kernel<1><<<M / 128, 256, SMEM, stream>>>(
          xi, xo, ladg, ctx16,
          even ? wE[s][0] : wO[s][0], even ? wE[s][1] : wO[s][1],
          even ? wE[s][2] : wO[s][2],
          even ? (eb0 + s * 128) : (ob0 + s * 128),
          even ? (eb1 + s * 128) : (ob1 + s * 128),
          even ? (eb2 + s * 46) : (ob2 + s * 23),
          an_ls, an_sh, li);
      float* t = xi; xi = xo; xo = t;
    }
  }
  sample_kernel<<<cdiv(M * 3, 256), 256, 0, stream>>>(
      xA, target, acc, sample, swb, twb);
  tv_kernel<<<cdiv(M * 3, 256), 256, 0, stream>>>(sample, acc);

  // ----- SSIM (separable 11-tap gaussian) -----
  for (int mode = 0; mode < 5; ++mode) {
    blur_h_kernel<<<cdiv(M * 3, 256), 256, 0, stream>>>(swb, twb, btmp, mode);
    blur_v_kernel<<<cdiv(M * 3, 256), 256, 0, stream>>>(btmp, blurOut[mode]);
  }
  ssim_kernel<<<cdiv(M * 3, 256), 256, 0, stream>>>(
      mpA, mtA, cpp, ctt, cpt, acc);

  // ----- radial power spectrum delta -----
  counts_kernel<<<cdiv(IMHW, 256), 256, 0, stream>>>(counts);
  dft_row_kernel<<<cdiv(24 * IMHW, 256), 256, 0, stream>>>(swb, twb, rowdft);
  dft_col_kernel<<<cdiv(24 * IMHW, 256), 256, 0, stream>>>(rowdft, binsum);
  pwr_kernel<<<1, 1, 0, stream>>>(binsum, counts, acc);

  final_kernel<<<1, 1, 0, stream>>>(acc, (float*)d_out);
}
